// FCOS_43688407335675
// MI455X (gfx1250) — compile-verified, matrix operand order checked by reading, not run on hardware
//
#include <hip/hip_runtime.h>
#include <math.h>

// ---------- types ----------
typedef __bf16 bf16_t;
typedef bf16_t bf16x16 __attribute__((ext_vector_type(16)));
typedef float  floatx8 __attribute__((ext_vector_type(8)));

union FragU { uint4 q[2]; bf16x16 v; };

__device__ __forceinline__ unsigned short f2bf(float f) {
  unsigned u = __float_as_uint(f);
  return (unsigned short)((u + 0x7FFFu + ((u >> 16) & 1u)) >> 16);
}
__device__ __forceinline__ float bf2f(unsigned short h) {
  return __uint_as_float(((unsigned)h) << 16);
}
__device__ __forceinline__ float sigm(float x) { return 1.0f / (1.0f + __expf(-x)); }

// ---------- weight packing: fp32 OIHW -> bf16 WMMA A-fragment order ----------
// idx = ((ks*nMfrag + mf)*32 + lane)*16 + e
//   m = mf*16 + (lane&15);  k = ks*32 + ((e<8)?e:e+8) + (lane>>4)*8
//   K order: tap-major, channel-minor: k = s*Cpad + ci
__global__ void fcos_pack_w(const float* __restrict__ w, unsigned short* __restrict__ wp,
                            int Cout, int Cin, int Cpad, int taps, int nMfrag, long total) {
  long idx = (long)blockIdx.x * blockDim.x + threadIdx.x;
  if (idx >= total) return;
  int e    = (int)(idx & 15);
  int lane = (int)((idx >> 4) & 31);
  long r   = idx >> 9;
  int mf   = (int)(r % nMfrag);
  int ks   = (int)(r / nMfrag);
  int m    = mf * 16 + (lane & 15);
  int k    = (ks << 5) + ((e < 8) ? e : (e + 8)) + (lane >> 4) * 8;
  int s    = k / Cpad;
  int ci   = k - s * Cpad;
  float v = 0.0f;
  if (m < Cout && ci < Cin) v = w[((size_t)m * Cin + ci) * taps + s];
  wp[idx] = f2bf(v);
}

// ---------- NCHW f32 -> NHWC bf16 (channel-padded with zeros) ----------
__global__ void fcos_cvt_nhwc(const float* __restrict__ src, unsigned short* __restrict__ dst,
                              int Cin, int Cpad, int HW, long total) {
  long i = (long)blockIdx.x * blockDim.x + threadIdx.x;
  if (i >= total) return;
  int c = (int)(i % Cpad);
  long t = i / Cpad;
  int p = (int)(t % HW);
  int b = (int)(t / HW);
  float v = (c < Cin) ? src[((size_t)b * Cin + c) * HW + p] : 0.0f;
  dst[i] = f2bf(v);
}

// ---------- NHWC: lat + nearest-up2(psmall), C = 256 ----------
__global__ void fcos_add_up2(const unsigned short* __restrict__ lat,
                             const unsigned short* __restrict__ psm,
                             unsigned short* __restrict__ outb,
                             int logW, long n) {
  long i = (long)blockIdx.x * blockDim.x + threadIdx.x;
  if (i >= n) return;
  const int W  = 1 << logW;
  const int HW = 1 << (2 * logW);
  int c = (int)(i & 255);
  long t = i >> 8;
  int p = (int)(t & (HW - 1));
  int b = (int)(t >> (2 * logW));
  int y = p >> logW, x = p & (W - 1);
  long si = (((long)b << (2 * logW - 2)) + ((long)(y >> 1) << (logW - 1)) + (x >> 1)) * 256 + c;
  outb[i] = f2bf(bf2f(lat[i]) + bf2f(psm[si]));
}

// ---------- implicit-GEMM conv via v_wmma_f32_16x16x32_bf16, NHWC ----------
// 128 threads = 4 wave32s; workgroup tile 64(Cout) x 128(pixels);
// wave tile 32x64 = 2(M) x 4(N) WMMA -> 8 accumulators, 8 WMMA per K-step.
// NHWC + K-map {g*8..+7, 16+g*8..+7} => B staging is pure b128 traffic.
// A pre-packed in fragment order => 2x global_load_b128 per fragment per lane.
// Epilogue: 8 contiguous channels per lane => 1x global_store_b128 per tile.
// NOTE: all 3x3 convs here have Cpad==256 (s = ks>>3); 1x1 convs have s==0.
__global__ __launch_bounds__(128) void fcos_conv_wmma(
    const unsigned short* __restrict__ act, const unsigned short* __restrict__ wpack,
    const float* __restrict__ bias, unsigned short* __restrict__ out,
    int Cpad, int Cout, int CsOut, int H, int W, int logW, int taps, int relu) {
  const int HW    = H * W;
  const int tid   = threadIdx.x;
  const int lane  = tid & 31;
  const int wv    = tid >> 5;
  const int wm    = wv >> 1;
  const int wn    = wv & 1;
  const int tileN = blockIdx.x;  // 128 pixels
  const int mblk  = blockIdx.y;  // 64 channels
  const int b     = blockIdx.z;
  const int nMfrag = gridDim.y * 4;
  const int nK     = (taps * Cpad) >> 5;

  __shared__ __align__(16) unsigned short Bsm[2][4096];  // 2 x 32(K) x 128(N) bf16

  // staging: 256 fragment slots (8 nfrags x 32 lanes), 2 per thread
  int st_py[2], st_px[2], st_pix[2], st_g[2];
#pragma unroll
  for (int h = 0; h < 2; ++h) {
    const int slot = tid + h * 128;
    const int sl   = slot & 31;
    st_g[h]   = sl >> 4;
    st_pix[h] = tileN * 128 + (slot >> 5) * 16 + (sl & 15);
    st_py[h]  = st_pix[h] >> logW;
    st_px[h]  = st_pix[h] & (W - 1);
  }
  const unsigned short* actb = act + (size_t)b * HW * Cpad;

  auto stage = [&](int ks, int buf) {
    int s, kb;
    if (taps == 9) { s = ks >> 3; kb = (ks & 7) << 5; }  // Cpad == 256
    else           { s = 0;       kb = ks << 5; }
    const int dy = (taps == 9) ? (s / 3 - 1) : 0;
    const int dx = (taps == 9) ? (s - (s / 3) * 3 - 1) : 0;
#pragma unroll
    for (int h = 0; h < 2; ++h) {
      const int sy = st_py[h] + dy;
      const int sx = st_px[h] + dx;
      const bool inb = (st_pix[h] < HW) & (sy >= 0) & (sy < H) & (sx >= 0) & (sx < W);
      uint4 q0 = make_uint4(0, 0, 0, 0), q1 = make_uint4(0, 0, 0, 0);
      if (inb) {
        const unsigned short* src = actb + (size_t)(sy * W + sx) * Cpad + kb + st_g[h] * 8;
        q0 = *(const uint4*)src;         // channels kb + g*8      .. +7
        q1 = *(const uint4*)(src + 16);  // channels kb + 16 + g*8 .. +7
      }
      uint4* dst = (uint4*)&Bsm[buf][(size_t)(tid + h * 128) * 16];
      dst[0] = q0;
      dst[1] = q1;
    }
  };

  floatx8 acc[2][4] = {};
  const int mf0 = mblk * 4 + wm * 2;
  stage(0, 0);
  for (int ks = 0; ks < nK; ++ks) {
    __syncthreads();
    if (ks + 1 < nK) {
      stage(ks + 1, (ks + 1) & 1);
      // prefetch next K-step's A fragments (global_prefetch_b8)
      __builtin_prefetch(wpack + ((size_t)((ks + 1) * nMfrag + mf0) * 32 + lane) * 16, 0, 3);
    }

    FragU a0, a1, bf[4];
    const uint4* ap0 = (const uint4*)(wpack + ((size_t)(ks * nMfrag + mf0)     * 32 + lane) * 16);
    const uint4* ap1 = (const uint4*)(wpack + ((size_t)(ks * nMfrag + mf0 + 1) * 32 + lane) * 16);
    a0.q[0] = ap0[0]; a0.q[1] = ap0[1];
    a1.q[0] = ap1[0]; a1.q[1] = ap1[1];
#pragma unroll
    for (int j = 0; j < 4; ++j) {
      const uint4* bp = (const uint4*)&Bsm[ks & 1][((wn * 4 + j) * 32 + lane) * 16];
      bf[j].q[0] = bp[0];
      bf[j].q[1] = bp[1];
    }
#pragma unroll
    for (int j = 0; j < 4; ++j) {
      acc[0][j] = __builtin_amdgcn_wmma_f32_16x16x32_bf16(false, a0.v, false, bf[j].v, (short)0, acc[0][j], false, false);
      acc[1][j] = __builtin_amdgcn_wmma_f32_16x16x32_bf16(false, a1.v, false, bf[j].v, (short)0, acc[1][j], false, false);
    }
  }

  // Epilogue: lane L -> N = L&15; VGPR v -> M = v + (L>>4)*8 (8 contiguous channels)
  const int g  = lane >> 4;
  const int nl = lane & 15;
#pragma unroll
  for (int i = 0; i < 2; ++i) {
    const int start = (mblk * 4 + wm * 2 + i) * 16 + g * 8;
#pragma unroll
    for (int j = 0; j < 4; ++j) {
      const int p = tileN * 128 + (wn * 4 + j) * 16 + nl;
      if (p < HW && start < CsOut) {
        union { unsigned short s16[8]; uint4 q; } o;
#pragma unroll
        for (int v = 0; v < 8; ++v) {
          const int m = start + v;
          float val = acc[i][j][v] + ((m < Cout) ? bias[m] : 0.0f);
          if (relu) val = fmaxf(val, 0.0f);
          o.s16[v] = f2bf(val);
        }
        *(uint4*)(out + ((size_t)b * HW + p) * CsOut + start) = o.q;
      }
    }
  }
}

// ---------- head: scores + decode (NHWC pred buffers, Cs = 32) ----------
__global__ void fcos_head(const unsigned short* cls3, const unsigned short* cls4, const unsigned short* cls5,
                          const unsigned short* box3, const unsigned short* box4, const unsigned short* box5,
                          const unsigned short* ctr3, const unsigned short* ctr4, const unsigned short* ctr5,
                          float* __restrict__ boxesAll, float* __restrict__ bestAll, int* __restrict__ clsAll) {
  int idx = blockIdx.x * blockDim.x + threadIdx.x;
  if (idx >= 8 * 5376) return;
  int b = idx / 5376, loc = idx - b * 5376;
  const unsigned short *cp, *bp, *tp;
  int HWl, logWl, i; float stride;
  if (loc < 4096)      { i = loc;        HWl = 4096; logWl = 6; stride =  8.0f; cp = cls3; bp = box3; tp = ctr3; }
  else if (loc < 5120) { i = loc - 4096; HWl = 1024; logWl = 5; stride = 16.0f; cp = cls4; bp = box4; tp = ctr4; }
  else                 { i = loc - 5120; HWl =  256; logWl = 4; stride = 32.0f; cp = cls5; bp = box5; tp = ctr5; }
  const size_t pb = ((size_t)b * HWl + i) * 32;
  float ctr = sigm(bf2f(tp[pb]));
  float best = -1.0f; int bc = 0;
  for (int c = 0; c < 20; ++c) {
    float pc = sigm(bf2f(cp[pb + c]));
    float s = sqrtf(pc * ctr);
    if (s > best) { best = s; bc = c; }
  }
  best = (best > 0.3f) ? best : 0.0f;
  int y = i >> logWl, x = i & ((1 << logWl) - 1);
  float xc = (x + 0.5f) * stride, yc = (y + 0.5f) * stride;
  float d0 = fmaxf(bf2f(bp[pb + 0]), 0.0f) * stride;
  float d1 = fmaxf(bf2f(bp[pb + 1]), 0.0f) * stride;
  float d2 = fmaxf(bf2f(bp[pb + 2]), 0.0f) * stride;
  float d3 = fmaxf(bf2f(bp[pb + 3]), 0.0f) * stride;
  float* bo = boxesAll + (size_t)idx * 4;
  bo[0] = fminf(fmaxf(xc - d0, 0.0f), 512.0f);
  bo[1] = fminf(fmaxf(yc - d1, 0.0f), 512.0f);
  bo[2] = fminf(fmaxf(xc + d2, 0.0f), 512.0f);
  bo[3] = fminf(fmaxf(yc + d3, 0.0f), 512.0f);
  bestAll[idx] = best;
  clsAll[idx]  = bc;
}

// ---------- per-batch bitonic top-1000 (full 8192-wide sort in LDS) ----------
__global__ __launch_bounds__(1024) void fcos_topk(const float* __restrict__ bestAll,
                                                  const float* __restrict__ boxesAll,
                                                  const int* __restrict__ clsAll,
                                                  float* __restrict__ tS, float* __restrict__ tB,
                                                  float* __restrict__ tC) {
  __shared__ float key[8192];
  __shared__ int   id[8192];
  const int b = blockIdx.x, tid = threadIdx.x;
  for (int t = 0; t < 8; ++t) {
    int i = t * 1024 + tid;
    key[i] = (i < 5376) ? bestAll[(size_t)b * 5376 + i] : -1e30f;
    id[i]  = (i < 5376) ? i : 0;
  }
  __syncthreads();
  for (int k = 2; k <= 8192; k <<= 1) {
    for (int j = k >> 1; j > 0; j >>= 1) {
      for (int t = 0; t < 8; ++t) {
        int i = t * 1024 + tid;
        int ixj = i ^ j;
        if (ixj > i) {
          bool desc = ((i & k) == 0);
          float ki = key[i], kj = key[ixj];
          bool sw = desc ? (ki < kj) : (ki > kj);
          if (sw) {
            key[i] = kj; key[ixj] = ki;
            int tmp = id[i]; id[i] = id[ixj]; id[ixj] = tmp;
          }
        }
      }
      __syncthreads();
    }
  }
  for (int t = tid; t < 1000; t += 1024) {
    int ii = id[t];
    tS[(size_t)b * 1000 + t] = key[t];
    tC[(size_t)b * 1000 + t] = (float)clsAll[(size_t)b * 5376 + ii];
    const float* src = boxesAll + ((size_t)b * 5376 + ii) * 4;
    float* dst = tB + ((size_t)b * 1000 + t) * 4;
    dst[0] = src[0]; dst[1] = src[1]; dst[2] = src[2]; dst[3] = src[3];
  }
}

// ---------- greedy class-offset NMS (matches reference fori_loop semantics) ----------
__global__ __launch_bounds__(1024) void fcos_nms(const float* __restrict__ tS, const float* __restrict__ tB,
                                                 const float* __restrict__ tC, float* __restrict__ outp) {
  const int b = blockIdx.x, tid = threadIdx.x;
  __shared__ float X1[1000], Y1[1000], X2[1000], Y2[1000], AR[1000], SC[1000];
  __shared__ int KEEP[1000];
  __shared__ int sup;
  for (int t = tid; t < 1000; t += 1024) {
    float off = tC[(size_t)b * 1000 + t] * 10000.0f;
    const float* bx = tB + ((size_t)b * 1000 + t) * 4;
    float x1 = bx[0] + off, y1 = bx[1] + off, x2 = bx[2] + off, y2 = bx[3] + off;
    X1[t] = x1; Y1[t] = y1; X2[t] = x2; Y2[t] = y2;
    AR[t] = (x2 - x1) * (y2 - y1);
    SC[t] = tS[(size_t)b * 1000 + t];
    KEEP[t] = 0;
  }
  __syncthreads();
  for (int i = 0; i < 1000; ++i) {
    if (tid == 0) sup = 0;
    __syncthreads();
    if (tid < 1000 && KEEP[tid]) {
      float ix1 = fmaxf(X1[i], X1[tid]), iy1 = fmaxf(Y1[i], Y1[tid]);
      float ix2 = fminf(X2[i], X2[tid]), iy2 = fminf(Y2[i], Y2[tid]);
      float inter = fmaxf(ix2 - ix1, 0.0f) * fmaxf(iy2 - iy1, 0.0f);
      float iou = inter / fmaxf(AR[i] + AR[tid] - inter, 1e-8f);
      if (iou > 0.5f) atomicOr(&sup, 1);
    }
    __syncthreads();
    if (tid == 0) KEEP[i] = (SC[i] > 0.0f) && (sup == 0);
    __syncthreads();
  }
  for (int t = tid; t < 1000; t += 1024) {
    float kf = KEEP[t] ? 1.0f : 0.0f;
    const float* bx = tB + ((size_t)b * 1000 + t) * 4;
    float* o = outp + ((size_t)b * 1000 + t) * 6;
    o[0] = bx[0] * kf; o[1] = bx[1] * kf; o[2] = bx[2] * kf; o[3] = bx[3] * kf;
    o[4] = tS[(size_t)b * 1000 + t] * kf;
    o[5] = tC[(size_t)b * 1000 + t] * kf;
  }
}

// =====================================================================
extern "C" void kernel_launch(void* const* d_in, const int* in_sizes, int n_in,
                              void* d_out, int out_size, void* d_ws, size_t ws_size,
                              hipStream_t stream) {
  (void)in_sizes; (void)n_in; (void)out_size; (void)ws_size;
  // d_in order (jax pytree: dict keys sorted): c3, c4, c5, then params:
  // lat3.b(3) lat3.w(4) lat4.b(5) lat4.w(6) lat5.b(7) lat5.w(8)
  // out3.b(9) out3.w(10) out4.b(11) out4.w(12) out5.b(13) out5.w(14)
  // pred_box.b(15) .w(16)  pred_cls.b(17) .w(18)  pred_ctr.b(19) .w(20)
  // stem_box[k]: b=21+2k, w=22+2k   stem_cls[k]: b=29+2k, w=30+2k
  const long HW3 = 4096, HW4 = 1024, HW5 = 256;
  char* ws = (char*)d_ws;
  size_t off = 0;
  auto alloc = [&](size_t bytes) { size_t o = off; off += (bytes + 255) & ~(size_t)255; return o; };
  auto U16 = [&](size_t o) { return (unsigned short*)(ws + o); };
  auto F32 = [&](size_t o) { return (float*)(ws + o); };
  auto I32 = [&](size_t o) { return (int*)(ws + o); };

  // bf16 NHWC activation buffers (channel stride = roundup(C,32))
  size_t o_c3b  = alloc(8L * HW3 * 64  * 2);
  size_t o_c4b  = alloc(8L * HW4 * 160 * 2);
  size_t o_c5b  = alloc(8L * HW5 * 416 * 2);
  size_t o_lat3 = alloc(8L * HW3 * 256 * 2);
  size_t o_lat4 = alloc(8L * HW4 * 256 * 2);
  size_t o_lat5 = alloc(8L * HW5 * 256 * 2);
  size_t o_sum3 = alloc(8L * HW3 * 256 * 2);
  size_t o_sum4 = alloc(8L * HW4 * 256 * 2);
  size_t o_p3   = alloc(8L * HW3 * 256 * 2);
  size_t o_p4   = alloc(8L * HW4 * 256 * 2);
  size_t o_p5   = alloc(8L * HW5 * 256 * 2);
  // stem ping/pong alias dead FPN buffers (lat3/sum3 free once p3 exists)
  size_t o_sA = o_sum3;
  size_t o_sB = o_lat3;
  // pred outputs (Cs = 32)
  size_t o_cls3 = alloc(8L * HW3 * 32 * 2), o_cls4 = alloc(8L * HW4 * 32 * 2), o_cls5 = alloc(8L * HW5 * 32 * 2);
  size_t o_box3 = alloc(8L * HW3 * 32 * 2), o_box4 = alloc(8L * HW4 * 32 * 2), o_box5 = alloc(8L * HW5 * 32 * 2);
  size_t o_ctr3 = alloc(8L * HW3 * 32 * 2), o_ctr4 = alloc(8L * HW4 * 32 * 2), o_ctr5 = alloc(8L * HW5 * 32 * 2);
  // f32 post buffers
  size_t o_boxA = alloc(8L * 5376 * 4 * 4);
  size_t o_best = alloc(8L * 5376 * 4);
  size_t o_clsA = alloc(8L * 5376 * 4);
  size_t o_tS = alloc(8L * 1000 * 4);
  size_t o_tC = alloc(8L * 1000 * 4);
  size_t o_tB = alloc(8L * 1000 * 4 * 4);

  auto packElems = [](int Cout, int Cin, int taps) -> long {
    long Cpad = (long)((Cin + 31) / 32) * 32;
    long Mpad = (long)((Cout + 63) / 64) * 64;
    return (taps * Cpad / 32) * (Mpad / 16) * 512L;
  };
  size_t o_wlat3 = alloc(packElems(256, 64, 1) * 2);
  size_t o_wlat4 = alloc(packElems(256, 160, 1) * 2);
  size_t o_wlat5 = alloc(packElems(256, 400, 1) * 2);
  size_t o_wout3 = alloc(packElems(256, 256, 9) * 2);
  size_t o_wout4 = alloc(packElems(256, 256, 9) * 2);
  size_t o_wout5 = alloc(packElems(256, 256, 9) * 2);
  size_t o_wsb[4], o_wsc[4];
  for (int k = 0; k < 4; ++k) o_wsb[k] = alloc(packElems(256, 256, 9) * 2);
  for (int k = 0; k < 4; ++k) o_wsc[k] = alloc(packElems(256, 256, 9) * 2);
  size_t o_wpc = alloc(packElems(20, 256, 9) * 2);
  size_t o_wpb = alloc(packElems(4, 256, 9) * 2);
  size_t o_wpt = alloc(packElems(1, 256, 9) * 2);

  auto packW = [&](int widx, size_t o, int Cout, int Cin, int taps) {
    int Cpad = (Cin + 31) / 32 * 32;
    int nMf = ((Cout + 63) / 64) * 4;
    long total = (long)(taps * Cpad / 32) * nMf * 512;
    fcos_pack_w<<<dim3((unsigned)((total + 255) / 256)), dim3(256), 0, stream>>>(
        (const float*)d_in[widx], U16(o), Cout, Cin, Cpad, taps, nMf, total);
  };
  packW(4, o_wlat3, 256, 64, 1);
  packW(6, o_wlat4, 256, 160, 1);
  packW(8, o_wlat5, 256, 400, 1);
  packW(10, o_wout3, 256, 256, 9);
  packW(12, o_wout4, 256, 256, 9);
  packW(14, o_wout5, 256, 256, 9);
  for (int k = 0; k < 4; ++k) packW(22 + 2 * k, o_wsb[k], 256, 256, 9);
  for (int k = 0; k < 4; ++k) packW(30 + 2 * k, o_wsc[k], 256, 256, 9);
  packW(18, o_wpc, 20, 256, 9);
  packW(16, o_wpb, 4, 256, 9);
  packW(20, o_wpt, 1, 256, 9);

  auto cvt = [&](int iidx, size_t o, int Cin, int Cpad, long HW) {
    long n = 8L * HW * Cpad;
    fcos_cvt_nhwc<<<dim3((unsigned)((n + 255) / 256)), dim3(256), 0, stream>>>(
        (const float*)d_in[iidx], U16(o), Cin, Cpad, (int)HW, n);
  };
  cvt(0, o_c3b, 64, 64, HW3);
  cvt(1, o_c4b, 160, 160, HW4);
  cvt(2, o_c5b, 400, 416, HW5);

  auto conv = [&](size_t inO, size_t wO, int biasIdx, size_t outO,
                  int Cin, int Cout, int taps, int W, int logW, int relu) {
    int Cpad  = (Cin + 31) / 32 * 32;
    int CsOut = (Cout + 31) / 32 * 32;
    int HW = W * W;
    dim3 grid((unsigned)((HW + 127) / 128), (unsigned)((Cout + 63) / 64), 8);
    fcos_conv_wmma<<<grid, dim3(128), 0, stream>>>(
        U16(inO), U16(wO), (const float*)d_in[biasIdx], U16(outO),
        Cpad, Cout, CsOut, W, W, logW, taps, relu);
  };

  // FPN
  conv(o_c3b, o_wlat3, 3, o_lat3, 64, 256, 1, 64, 6, 0);
  conv(o_c4b, o_wlat4, 5, o_lat4, 160, 256, 1, 32, 5, 0);
  conv(o_c5b, o_wlat5, 7, o_lat5, 400, 256, 1, 16, 4, 0);
  conv(o_lat5, o_wout5, 13, o_p5, 256, 256, 9, 16, 4, 0);
  {
    long n = 8L * HW4 * 256;
    fcos_add_up2<<<dim3((unsigned)((n + 255) / 256)), dim3(256), 0, stream>>>(
        U16(o_lat4), U16(o_p5), U16(o_sum4), 5, n);
  }
  conv(o_sum4, o_wout4, 11, o_p4, 256, 256, 9, 32, 5, 0);
  {
    long n = 8L * HW3 * 256;
    fcos_add_up2<<<dim3((unsigned)((n + 255) / 256)), dim3(256), 0, stream>>>(
        U16(o_lat3), U16(o_p4), U16(o_sum3), 6, n);
  }
  conv(o_sum3, o_wout3, 9, o_p3, 256, 256, 9, 64, 6, 0);

  // per-level heads
  const size_t pO[3]    = {o_p3, o_p4, o_p5};
  const int    Ws[3]    = {64, 32, 16};
  const int    logWs[3] = {6, 5, 4};
  const size_t clsO[3]  = {o_cls3, o_cls4, o_cls5};
  const size_t boxO[3]  = {o_box3, o_box4, o_box5};
  const size_t ctrO[3]  = {o_ctr3, o_ctr4, o_ctr5};
  for (int L = 0; L < 3; ++L) {
    int W = Ws[L], logW = logWs[L];
    // cls stem (sA/sB alias lat3/sum3, each larger than any level's activation)
    size_t cur = pO[L];
    for (int s = 0; s < 4; ++s) {
      size_t nxt = (s & 1) ? o_sB : o_sA;
      conv(cur, o_wsc[s], 29 + 2 * s, nxt, 256, 256, 9, W, logW, 1);
      cur = nxt;
    }
    conv(cur, o_wpc, 17, clsO[L], 256, 20, 9, W, logW, 0);
    // box stem
    cur = pO[L];
    for (int s = 0; s < 4; ++s) {
      size_t nxt = (s & 1) ? o_sB : o_sA;
      conv(cur, o_wsb[s], 21 + 2 * s, nxt, 256, 256, 9, W, logW, 1);
      cur = nxt;
    }
    conv(cur, o_wpb, 15, boxO[L], 256, 4, 9, W, logW, 0);
    conv(cur, o_wpt, 19, ctrO[L], 256, 1, 9, W, logW, 0);
  }

  // scores + decode
  {
    long n = 8L * 5376;
    fcos_head<<<dim3((unsigned)((n + 255) / 256)), dim3(256), 0, stream>>>(
        U16(o_cls3), U16(o_cls4), U16(o_cls5),
        U16(o_box3), U16(o_box4), U16(o_box5),
        U16(o_ctr3), U16(o_ctr4), U16(o_ctr5),
        F32(o_boxA), F32(o_best), I32(o_clsA));
  }
  // top-k
  fcos_topk<<<dim3(8), dim3(1024), 0, stream>>>(F32(o_best), F32(o_boxA), I32(o_clsA),
                                                F32(o_tS), F32(o_tB), F32(o_tC));
  // NMS + final output
  fcos_nms<<<dim3(8), dim3(1024), 0, stream>>>(F32(o_tS), F32(o_tB), F32(o_tC), (float*)d_out);
}
